// CausalSelfAttention_21895743275381
// MI455X (gfx1250) — compile-verified, matrix operand order checked by reading, not run on hardware
//
#include <hip/hip_runtime.h>
#include <hip/hip_bf16.h>

// ---------------------------------------------------------------------------
// Causal self-attention (RoPE) for MI455X / gfx1250, wave32 + WMMA f16.
//
//   1) cvt: x, Wq, Wk, Wv, Wp  -> f16 workspace copies
//   2) gemm_wmma: LDS-staged block GEMM (128x64 block, 8 waves, 32x32/wave),
//      double-buffered K-tiles copied with GLOBAL_LOAD_ASYNC_TO_LDS_B128
//      (ASYNCcnt); q,k -> (B,H,T,64), v -> (B,H,64,T).
//   3) rope_qk: in-place rotary embedding on q,k
//   4) flash_attn: one wave per (b,h,16-row q tile), S^T formulation,
//      online softmax, O^T += V^T @ P^T, LDS relayout for P (s_wait_dscnt).
//   5) gemm_wmma (mode 2): out = y@Wp^T + bp -> f32 d_out
// ---------------------------------------------------------------------------

typedef __attribute__((ext_vector_type(16))) _Float16 v16h;
typedef __attribute__((ext_vector_type(8)))  _Float16 v8h;
typedef __attribute__((ext_vector_type(8)))  float    v8f;
typedef __attribute__((ext_vector_type(4)))  int      v4i;

#define LOG2E 1.4426950408889634f

#if __has_builtin(__builtin_amdgcn_global_load_async_to_lds_b128)
#define HAS_ASYNC_LDS 1
#else
#define HAS_ASYNC_LDS 0
#endif

typedef __attribute__((address_space(1))) v4i* gv4i_p;
typedef __attribute__((address_space(3))) v4i* lv4i_p;

__device__ __forceinline__ v8f wmma_f16(v16h a, v16h b, v8f c) {
  return __builtin_amdgcn_wmma_f32_16x16x32_f16(false, a, false, b,
                                                (short)0, c, false, false);
}

// 16-byte global -> LDS copy (async on CDNA5, sync fallback otherwise)
__device__ __forceinline__ void cp16(const _Float16* g, _Float16* l) {
#if HAS_ASYNC_LDS
  __builtin_amdgcn_global_load_async_to_lds_b128((gv4i_p)(_Float16*)g,
                                                 (lv4i_p)l, 0, 0);
#else
  *(v8h*)l = *(const v8h*)g;
#endif
}

__device__ __forceinline__ void wait_async_copies() {
#if HAS_ASYNC_LDS
#if __has_builtin(__builtin_amdgcn_s_wait_asynccnt)
  __builtin_amdgcn_s_wait_asynccnt(0);
#else
  asm volatile("s_wait_asynccnt 0" ::: "memory");
#endif
#endif
}

__device__ __forceinline__ v16h ldsv16(const _Float16* p) {
  v8h lo = *(const v8h*)p;
  v8h hi = *(const v8h*)(p + 8);
  return __builtin_shufflevector(lo, hi, 0, 1, 2, 3, 4, 5, 6, 7,
                                 8, 9, 10, 11, 12, 13, 14, 15);
}

// ---------------------------------------------------------------- cvt f32->f16
__global__ void cvt_f32_f16(const float* __restrict__ in,
                            _Float16* __restrict__ out, int n) {
  int i = blockIdx.x * blockDim.x + threadIdx.x;
  if (i < n) out[i] = (_Float16)in[i];
}

// ------------------------------------------------------------------- RoPE q,k
__global__ void rope_qk(_Float16* __restrict__ q, _Float16* __restrict__ k,
                        const float* __restrict__ rope) {
  int i = blockIdx.x * blockDim.x + threadIdx.x;
  int p  = i & 31;
  int t  = (i >> 5) & 2047;
  int bh = i >> 16;                       // 32*2048 = 2^16
  size_t base = ((size_t)bh * 2048 + t) * 64 + 2 * p;
  float c = rope[(t * 32 + p) * 2];
  float s = rope[(t * 32 + p) * 2 + 1];
  float q0 = (float)q[base], q1 = (float)q[base + 1];
  q[base]     = (_Float16)(q0 * c - q1 * s);
  q[base + 1] = (_Float16)(q0 * s + q1 * c);
  float k0 = (float)k[base], k1 = (float)k[base + 1];
  k[base]     = (_Float16)(k0 * c - k1 * s);
  k[base + 1] = (_Float16)(k0 * s + k1 * c);
}

// ----------------------------------------------- LDS-staged block WMMA GEMM
// out = A(M x K) @ W(N x K)^T + bias.   M=4096, N=K=1024.
// Block: 256 thr = 8 waves; block tile 128(M) x 64(N); wave tile 32x32.
// K staged in double-buffered LDS (A:128x32, B:64x32, pad to 40 halves/row).
// mode 0: f16 -> (B,H,T,64)   mode 1: f16 -> (B,H,64,T)   mode 2: f32 M x N
#define GSTRIDE 40
__global__ __launch_bounds__(256)
void gemm_wmma(const _Float16* __restrict__ A, const _Float16* __restrict__ W,
               const float* __restrict__ bias, void* __restrict__ dst, int mode) {
  __shared__ _Float16 As[2][128 * GSTRIDE];
  __shared__ _Float16 Bs[2][64 * GSTRIDE];
  const int LD = 1024;
  int tid = threadIdx.x;
  int lane = tid & 31, ln = lane & 15, hf = lane >> 4;
  int w = tid >> 5, wm = w & 3, wn = w >> 2;       // 4(M) x 2(N) waves
  int m0 = blockIdx.y * 128, n0 = blockIdx.x * 64;

  // copy-index mapping (whole block copies one K-step tile pair)
  int ar = tid >> 1, ac = (tid & 1) * 16;          // A: 128 rows x 32 halves
  int br = tid >> 2, bc = (tid & 3) * 8;           // B:  64 rows x 32 halves
  const _Float16* agp = A + (size_t)(m0 + ar) * LD + ac;
  const _Float16* bgp = W + (size_t)(n0 + br) * LD + bc;
  _Float16* alp[2] = { As[0] + ar * GSTRIDE + ac, As[1] + ar * GSTRIDE + ac };
  _Float16* blp[2] = { Bs[0] + br * GSTRIDE + bc, Bs[1] + br * GSTRIDE + bc };

  // prologue: stage K-step 0 into buffer 0
  cp16(agp, alp[0]);
  cp16(agp + 8, alp[0] + 8);
  cp16(bgp, blp[0]);
  wait_async_copies();
  __syncthreads();

  v8f acc[2][2] = {};
  int cur = 0;
  for (int step = 0; step < 32; ++step) {
    if (step + 1 < 32) {                 // stage next K-step into other buffer
      int k = (step + 1) * 32;
      cp16(agp + k, alp[cur ^ 1]);
      cp16(agp + k + 8, alp[cur ^ 1] + 8);
      cp16(bgp + k, blp[cur ^ 1]);
    }
    const _Float16* Ab = As[cur] + (wm * 32 + ln) * GSTRIDE + hf * 16;
    const _Float16* Bb = Bs[cur] + (wn * 32 + ln) * GSTRIDE + hf * 16;
    v16h a0 = ldsv16(Ab);
    v16h a1 = ldsv16(Ab + 16 * GSTRIDE);
    v16h b0 = ldsv16(Bb);
    v16h b1 = ldsv16(Bb + 16 * GSTRIDE);
    acc[0][0] = wmma_f16(a0, b0, acc[0][0]);
    acc[0][1] = wmma_f16(a0, b1, acc[0][1]);
    acc[1][0] = wmma_f16(a1, b0, acc[1][0]);
    acc[1][1] = wmma_f16(a1, b1, acc[1][1]);
    wait_async_copies();                 // next-buffer copies landed
    __syncthreads();                     // everyone done reading cur
    cur ^= 1;
  }

  int m0w = m0 + wm * 32, n0w = n0 + wn * 32;
  float bn[2] = { bias[n0w + ln], bias[n0w + 16 + ln] };
#pragma unroll
  for (int mi = 0; mi < 2; ++mi)
#pragma unroll
    for (int ni = 0; ni < 2; ++ni) {
      int n = n0w + ni * 16 + ln;
#pragma unroll
      for (int j = 0; j < 8; ++j) {
        int m = m0w + mi * 16 + hf * 8 + j;
        float v = acc[mi][ni][j] + bn[ni];
        if (mode == 2) {
          ((float*)dst)[(size_t)m * LD + n] = v;
        } else {
          int b = m >> 11, t = m & 2047, h = n >> 6, d = n & 63;
          _Float16 hv = (_Float16)v;
          if (mode == 0)
            ((_Float16*)dst)[(((size_t)(b * 16 + h)) * 2048 + t) * 64 + d] = hv;
          else
            ((_Float16*)dst)[(((size_t)(b * 16 + h)) * 64 + d) * 2048 + t] = hv;
        }
      }
    }
}

// --------------------------------------------------------- flash attention
__global__ __launch_bounds__(32)
void flash_attn(const _Float16* __restrict__ qh, const _Float16* __restrict__ kh,
                const _Float16* __restrict__ vT, _Float16* __restrict__ yh) {
  __shared__ _Float16 ldsP[16 * 40];   // 16 t-rows, 32 s + pad (bank spread)
  __shared__ _Float16 ldsY[16 * 72];   // 16 t-rows, 64 d + pad
  int lane = threadIdx.x, ln = lane & 15, hf = lane >> 4;
  int t0 = blockIdx.x * 16;
  int h  = blockIdx.y;
  int b  = blockIdx.z;
  int bh = b * 16 + h;
  const _Float16* qb = qh + ((size_t)bh * 2048) * 64;
  const _Float16* kb = kh + ((size_t)bh * 2048) * 64;
  const _Float16* vb = vT + ((size_t)bh * 64) * 2048;

  v16h bq0 = *(const v16h*)(qb + (size_t)(t0 + ln) * 64 + hf * 16);
  v16h bq1 = *(const v16h*)(qb + (size_t)(t0 + ln) * 64 + 32 + hf * 16);

  v8f acc[4] = {};
  float m_i = -3.0e38f, l_i = 0.0f;
  int tg = t0 + ln;

  for (int s0 = 0; s0 < t0 + 16; s0 += 32) {
    v16h ka00 = *(const v16h*)(kb + (size_t)(s0 + ln) * 64 + hf * 16);
    v16h ka01 = *(const v16h*)(kb + (size_t)(s0 + ln) * 64 + 32 + hf * 16);
    v16h ka10 = *(const v16h*)(kb + (size_t)(s0 + 16 + ln) * 64 + hf * 16);
    v16h ka11 = *(const v16h*)(kb + (size_t)(s0 + 16 + ln) * 64 + 32 + hf * 16);
    v8f st0 = {}, st1 = {};
    st0 = wmma_f16(ka00, bq0, st0);
    st0 = wmma_f16(ka01, bq1, st0);
    st1 = wmma_f16(ka10, bq0, st1);
    st1 = wmma_f16(ka11, bq1, st1);

    float vals[16];
    float mloc = -3.0e38f;
#pragma unroll
    for (int r = 0; r < 2; ++r)
#pragma unroll
      for (int j = 0; j < 8; ++j) {
        int sg = s0 + r * 16 + hf * 8 + j;
        float v = (r ? st1[j] : st0[j]) * 0.125f;   // 1/sqrt(64)
        if (sg > tg) v = -1.0e30f;
        vals[r * 8 + j] = v;
        mloc = fmaxf(mloc, v);
      }
    mloc = fmaxf(mloc, __shfl_xor(mloc, 16, 32));
    float m_new = fmaxf(m_i, mloc);

    float ssum = 0.0f;
    _Float16 ph[16];
#pragma unroll
    for (int e = 0; e < 16; ++e) {
      float p = exp2f((vals[e] - m_new) * LOG2E);
      ssum += p;
      ph[e] = (_Float16)p;
    }
    ssum += __shfl_xor(ssum, 16, 32);
    float alpha = exp2f((m_i - m_new) * LOG2E);
    l_i = l_i * alpha + ssum;
    m_i = m_new;
#pragma unroll
    for (int d = 0; d < 4; ++d)
#pragma unroll
      for (int j = 0; j < 8; ++j) acc[d][j] *= alpha;

#pragma unroll
    for (int r = 0; r < 2; ++r) {
      v8h pk;
#pragma unroll
      for (int j = 0; j < 8; ++j) pk[j] = ph[r * 8 + j];
      *(v8h*)(ldsP + ln * 40 + r * 16 + hf * 8) = pk;
    }
    asm volatile("s_wait_dscnt 0" ::: "memory");
    v8h plo = *(const v8h*)(ldsP + ln * 40 + hf * 16);
    v8h phi = *(const v8h*)(ldsP + ln * 40 + hf * 16 + 8);
    v16h bp = __builtin_shufflevector(plo, phi, 0, 1, 2, 3, 4, 5, 6, 7,
                                      8, 9, 10, 11, 12, 13, 14, 15);

#pragma unroll
    for (int d = 0; d < 4; ++d) {
      v16h va = *(const v16h*)(vb + ((size_t)(d * 16 + ln)) * 2048 + s0 + hf * 16);
      acc[d] = wmma_f16(va, bp, acc[d]);
    }
  }

  float rl = 1.0f / l_i;
#pragma unroll
  for (int d = 0; d < 4; ++d) {
    v8h yk;
#pragma unroll
    for (int j = 0; j < 8; ++j) yk[j] = (_Float16)(acc[d][j] * rl);
    *(v8h*)(ldsY + ln * 72 + d * 16 + hf * 8) = yk;
  }
  asm volatile("s_wait_dscnt 0" ::: "memory");
  size_t orow = ((size_t)(b * 2048 + t0 + ln)) * 1024 + h * 64 + hf * 32;
#pragma unroll
  for (int c = 0; c < 4; ++c) {
    v8h yv = *(const v8h*)(ldsY + ln * 72 + hf * 32 + c * 8);
    *(v8h*)(yh + orow + c * 8) = yv;
  }
}

// ---------------------------------------------------------------------------
extern "C" void kernel_launch(void* const* d_in, const int* in_sizes, int n_in,
                              void* d_out, int out_size, void* d_ws, size_t ws_size,
                              hipStream_t stream) {
  const float* x    = (const float*)d_in[0];
  const float* Wq   = (const float*)d_in[1];
  const float* bq   = (const float*)d_in[2];
  const float* Wk   = (const float*)d_in[3];
  const float* bk   = (const float*)d_in[4];
  const float* Wv   = (const float*)d_in[5];
  const float* bv   = (const float*)d_in[6];
  const float* Wp   = (const float*)d_in[7];
  const float* bp   = (const float*)d_in[8];
  const float* rope = (const float*)d_in[9];
  float* out = (float*)d_out;

  char* w = (char*)d_ws;
  const size_t MB = 1024 * 1024;
  _Float16* xh  = (_Float16*)(w + 0 * MB);   // 4096x1024
  _Float16* wqh = (_Float16*)(w + 8 * MB);   // 1024x1024 each
  _Float16* wkh = (_Float16*)(w + 16 * MB);
  _Float16* wvh = (_Float16*)(w + 24 * MB);
  _Float16* wph = (_Float16*)(w + 32 * MB);
  _Float16* qh  = (_Float16*)(w + 40 * MB);  // (B,H,T,64)
  _Float16* kh  = (_Float16*)(w + 48 * MB);  // (B,H,T,64)
  _Float16* vt  = (_Float16*)(w + 56 * MB);  // (B,H,64,T)
  _Float16* yh  = (_Float16*)(w + 64 * MB);  // (B,T,C)

  const int NX = 4096 * 1024, NW = 1024 * 1024;
  cvt_f32_f16<<<NX / 256, 256, 0, stream>>>(x, xh, NX);
  cvt_f32_f16<<<NW / 256, 256, 0, stream>>>(Wq, wqh, NW);
  cvt_f32_f16<<<NW / 256, 256, 0, stream>>>(Wk, wkh, NW);
  cvt_f32_f16<<<NW / 256, 256, 0, stream>>>(Wv, wvh, NW);
  cvt_f32_f16<<<NW / 256, 256, 0, stream>>>(Wp, wph, NW);

  dim3 gg(1024 / 64, 4096 / 128);            // 16 x 32 blocks of 8 waves
  gemm_wmma<<<gg, 256, 0, stream>>>(xh, wqh, bq, (void*)qh, 0);
  gemm_wmma<<<gg, 256, 0, stream>>>(xh, wkh, bk, (void*)kh, 0);
  gemm_wmma<<<gg, 256, 0, stream>>>(xh, wvh, bv, (void*)vt, 1);

  rope_qk<<<(2 * 16 * 2048 * 32) / 256, 256, 0, stream>>>(qh, kh, rope);

  flash_attn<<<dim3(128, 16, 2), 32, 0, stream>>>(qh, kh, vt, yh);

  gemm_wmma<<<gg, 256, 0, stream>>>(yh, wph, bp, (void*)out, 2);
}
